// TGAT_78357383348807
// MI455X (gfx1250) — matrix operand order repaired and checked by reading, not compiled, not run on hardware
//
#include <hip/hip_runtime.h>
#include <hip/hip_bf16.h>
#include <stdint.h>

// ---------------------------------------------------------------------------
// CDNA5 (gfx1250) fused TGAT forward.
// Math rewrites: constant query vector -> scores via 2-col matvec (wk_eff);
// attention pooling before the V projection (linearity). All dense math on
// v_wmma_f32_16x16x32_bf16 (f32 accumulate), data movement bf16.
// ---------------------------------------------------------------------------

typedef __attribute__((ext_vector_type(16))) __bf16 v16bf;
typedef __attribute__((ext_vector_type(8)))  float  v8f;

#define TD   100   // TIME_DIM
#define NB   16    // neighbors per node
#define QDIM 228
#define HDIM 114

__device__ __forceinline__ unsigned short f2bf(float f) {
  unsigned int u = __float_as_uint(f);
  u += 0x7FFFu + ((u >> 16) & 1u);          // round-to-nearest-even
  return (unsigned short)(u >> 16);
}
__device__ __forceinline__ float bf2f(unsigned short h) {
  return __uint_as_float(((unsigned int)h) << 16);
}

struct FragBits { uint4 lo, hi; };
__device__ __forceinline__ v16bf frag_from(uint4 lo, uint4 hi) {
  union { FragBits b; v16bf v; } u;
  u.b.lo = lo; u.b.hi = hi;
  return u.v;
}

// A-fragment (16x32 bf16, M x K) from LDS row-major [row*stride + k].
// ISA 7.12.2: lanes 0-15 -> row M=lane, K {0..7,16..23}; lanes 16-31 -> K {8..15,24..31}.
__device__ __forceinline__ v16bf lds_a_frag(const unsigned short* base, int mbase,
                                            int kbase, int stride) {
  const int lane = threadIdx.x & 31;
  const int sel  = lane >> 4;
  const unsigned short* p = base + (mbase + (lane & 15)) * stride + kbase + sel * 8;
  return frag_from(*(const uint4*)p, *(const uint4*)(p + 16));
}

// B-fragment (32x16 bf16, K x N) from global weights stored row-major by n:
// W[n*kstride + k] (B[k][n] = W[n][k]).  lanes 0-15: N=lane, K {0..7,16..23};
// lanes 16-31: same N, K {8..15,24..31}  => two 16B loads per lane (matches A).
__device__ __forceinline__ v16bf glb_b_frag(const unsigned short* __restrict__ W,
                                            int nbase, int kbase, int kstride) {
  const int lane = threadIdx.x & 31;
  const unsigned short* p = W + (nbase + (lane & 15)) * kstride + kbase + (lane >> 4) * 8;
  return frag_from(*(const uint4*)p, *(const uint4*)(p + 16));
}

__device__ __forceinline__ v8f wmma_bf16(v16bf a, v16bf b, v8f c) {
  return __builtin_amdgcn_wmma_f32_16x16x32_bf16(false, a, false, b, (short)0, c,
                                                 false, false);
}

// ---------------------------------------------------------------------------
// Prep: cos(time_b), constant q per layer, q.bk, wk_eff packed bf16
// ---------------------------------------------------------------------------
__global__ __launch_bounds__(256) void prep1_kernel(
    const float* __restrict__ time_w, const float* __restrict__ time_b,
    const float* __restrict__ Wq0, const float* __restrict__ bq0,
    const float* __restrict__ Wk0, const float* __restrict__ bk0,
    const float* __restrict__ Wq1, const float* __restrict__ bq1,
    const float* __restrict__ Wk1, const float* __restrict__ bk1,
    float* __restrict__ qbk_out,                 // [layer*2 + head]
    unsigned short* __restrict__ wkB0,           // 16 x 320
    unsigned short* __restrict__ wkB1)           // 16 x 192
{
  __shared__ float cb[TD];
  __shared__ float qv[2][QDIM];
  const int t = threadIdx.x;
  (void)time_w;
  if (t < TD) cb[t] = __cosf(time_b[t]);        // time2vec at t=0 -> cos(b)
  __syncthreads();
  for (int i = t; i < 2 * QDIM; i += 256) {
    const int l = i / QDIM, n = i % QDIM;
    const float* Wq = l ? Wq1 : Wq0;
    const float* bq = l ? bq1 : bq0;
    float s = bq[n];
    for (int d = 0; d < TD; ++d) s = fmaf(Wq[n * QDIM + 128 + d], cb[d], s);
    qv[l][n] = s;
  }
  __syncthreads();
  if (t < 4) {
    const int l = t >> 1, h = t & 1;
    const float* bk = l ? bk1 : bk0;
    float s = 0.f;
    for (int d = 0; d < HDIM; ++d) s = fmaf(qv[l][h * HDIM + d], bk[h * HDIM + d], s);
    qbk_out[t] = s;
  }
  // wk_eff layer0: kv cols are identity 0..291 of Wk0 (KD=292), Kpad=320
  for (int i = t; i < 16 * 320; i += 256) {
    const int r = i / 320, c = i % 320;
    unsigned short v = 0;
    if (r < 2 && c < 292) {
      float s = 0.f;
      for (int d = 0; d < HDIM; ++d)
        s = fmaf(qv[0][r * HDIM + d], Wk0[(r * HDIM + d) * 292 + c], s);
      v = f2bf(s);
    }
    wkB0[i] = v;
  }
  // wk_eff layer1: kv cols map to Wk1 cols 128..291 (nbr feats are zero), Kpad=192
  for (int i = t; i < 16 * 192; i += 256) {
    const int r = i / 192, c = i % 192;
    unsigned short v = 0;
    if (r < 2 && c < 164) {
      float s = 0.f;
      for (int d = 0; d < HDIM; ++d)
        s = fmaf(qv[1][r * HDIM + d], Wk1[(r * HDIM + d) * 292 + 128 + c], s);
      v = f2bf(s);
    }
    wkB1[i] = v;
  }
}

// Generic f32 -> bf16 weight repack into B-fragment friendly row-major (n, k).
__global__ __launch_bounds__(256) void pack_w_kernel(
    const float* __restrict__ src, unsigned short* __restrict__ dst,
    int n_src, int k_src_stride, int k_off, int k_width, int N_dst, int K_dst)
{
  const int i = blockIdx.x * 256 + threadIdx.x;
  if (i >= N_dst * K_dst) return;
  const int n = i / K_dst, c = i % K_dst;
  unsigned short v = 0;
  if (n < n_src && c < k_width) v = f2bf(src[n * k_src_stride + k_off + c]);
  dst[i] = v;
}

__global__ __launch_bounds__(256) void zero_f4_kernel(float4* __restrict__ p, int n4) {
  const int i = blockIdx.x * 256 + threadIdx.x;
  if (i < n4) p[i] = make_float4(0.f, 0.f, 0.f, 0.f);
}

__global__ __launch_bounds__(256) void scatter_rows_kernel(
    const float* __restrict__ src, float* __restrict__ z,
    const int* __restrict__ nids, const int* __restrict__ n2i, int nrows)
{
  const int i = blockIdx.x * 256 + threadIdx.x;
  if (i >= nrows * 128) return;
  const int r = i >> 7, c = i & 127;
  z[(size_t)n2i[nids[r]] * 128 + c] = src[i];
}

// ---------------------------------------------------------------------------
// Fused attention layer. One WG = NODES nodes (M = NODES*16 kv rows), 8 waves.
// ---------------------------------------------------------------------------
template <int NODES, int KPAD, int KVW, bool GATHER>
__global__ __launch_bounds__(256, 1) void attn_fused_kernel(
    const float* __restrict__ edge_feats,   // (N, 16, 64)
    const float* __restrict__ times,        // (N)
    const float* __restrict__ nbr_times,    // (N, 16)
    const int*   __restrict__ nbr_idx,      // (N, 16) or null
    const float* __restrict__ zsrc,         // (U, 128) gather source
    const int*   __restrict__ nids,         // scatter ids or null
    const int*   __restrict__ n2i,
    const float* __restrict__ timew, const float* __restrict__ timeb,
    const unsigned short* __restrict__ WvB,   // 240 x KPAD
    const unsigned short* __restrict__ wkB,   // 16  x KPAD
    const unsigned short* __restrict__ WoB,   // 240 x 256
    const unsigned short* __restrict__ Wm1B,  // 128 x 256
    const unsigned short* __restrict__ Wm2B,  // 128 x 128
    const float* __restrict__ qbk2,           // 2 floats (per head)
    const float* __restrict__ bv, const float* __restrict__ bo,
    const float* __restrict__ bm1, const float* __restrict__ bm2,
    float* __restrict__ dst)                  // (.,128)
{
  constexpr int M   = NODES * 16;       // kv rows
  constexpr int PM  = NODES * 2;        // pooled rows (node,head)
  constexpr int PMT = PM / 16 ? PM / 16 : 1;
  constexpr int KS  = KPAD + 8;         // LDS row stride (halves)
  constexpr int KT  = KPAD / 32;

  __shared__ unsigned short s_kv[M * KS];
  __shared__ float          s_sc[M * 2];
  __shared__ float          s_attn[PM * NB];
  __shared__ unsigned short s_pool[PM * KS];
  __shared__ unsigned short s_out[16 * 264];
  __shared__ unsigned short s_o2 [16 * 264];
  __shared__ unsigned short s_h  [16 * 136];
  __shared__ float          s_tw[TD], s_tb[TD];

  const int t    = threadIdx.x;
  const int wv   = t >> 5;
  const int lane = t & 31;
  const int sel  = lane >> 4;
  const int ln   = lane & 15;
  const int node0 = blockIdx.x * NODES;

  if (t < TD) s_tw[t] = timew[t];
  else if (t < 2 * TD) s_tb[t - TD] = timeb[t - TD];
  { // zero out columns 240..255 of the 240-wide epilogue buffers
    const int r = t >> 4, c = 240 + (t & 15);
    s_out[r * 264 + c] = 0;
    s_o2 [r * 264 + c] = 0;
  }
  __syncthreads();

  // ---- Phase A: build kv rows (bf16) in LDS ----
  if (t < M) {
    const int node = node0 + (t >> 4);
    const int kk   = t & 15;
    unsigned short* row = s_kv + t * KS;
    int c = 0;
    if constexpr (GATHER) {
      const int gi = nbr_idx[node * NB + kk];
      const float4* zp = (const float4*)(zsrc + (size_t)gi * 128);
#pragma unroll
      for (int i = 0; i < 32; ++i) {
        float4 v = zp[i];
        row[c] = f2bf(v.x); row[c + 1] = f2bf(v.y);
        row[c + 2] = f2bf(v.z); row[c + 3] = f2bf(v.w);
        c += 4;
      }
    }
    const float4* ep = (const float4*)(edge_feats + ((size_t)node * NB + kk) * 64);
#pragma unroll
    for (int i = 0; i < 16; ++i) {
      float4 v = ep[i];
      row[c] = f2bf(v.x); row[c + 1] = f2bf(v.y);
      row[c + 2] = f2bf(v.z); row[c + 3] = f2bf(v.w);
      c += 4;
    }
    const float dt = times[node] - nbr_times[node * NB + kk];
#pragma unroll 4
    for (int d = 0; d < TD; ++d)
      row[c + d] = f2bf(__cosf(fmaf(dt, s_tw[d], s_tb[d])));
    c += TD;
    for (; c < KPAD; ++c) row[c] = 0;   // (KVW..KPAD) zero pad
  }
  __syncthreads();

  // ---- Phase B: scores = kv . wk_eff  (WMMA, only N cols 0,1 meaningful) ----
  const float inv_sqrt_hd = 0.09366863f;  // 1/sqrt(114)
  const float qb = (ln < 2) ? qbk2[ln] : 0.f;   // hoisted out of the mt loop
  for (int mt = wv; mt < NODES; mt += 8) {
    v8f acc = {0.f, 0.f, 0.f, 0.f, 0.f, 0.f, 0.f, 0.f};
#pragma unroll
    for (int kt = 0; kt < KT; ++kt) {
      v16bf a = lds_a_frag(s_kv, mt * 16, kt * 32, KS);
      v16bf b = glb_b_frag(wkB, 0, kt * 32, KPAD);
      acc = wmma_bf16(a, b, acc);
    }
    if (ln < 2) {
#pragma unroll
      for (int i = 0; i < 8; ++i)
        s_sc[(mt * 16 + i + 8 * sel) * 2 + ln] = (acc[i] + qb) * inv_sqrt_hd;
    }
  }
  __syncthreads();

  // ---- Phase C: softmax over K=16 per (node, head), then kv pooling ----
  if (t < PM) {
    const int node = t >> 1, h = t & 1;
    float sc[NB], mx = -1e30f;
#pragma unroll
    for (int k = 0; k < NB; ++k) {
      sc[k] = s_sc[(node * NB + k) * 2 + h];
      mx = fmaxf(mx, sc[k]);
    }
    float sum = 0.f;
#pragma unroll
    for (int k = 0; k < NB; ++k) { sc[k] = __expf(sc[k] - mx); sum += sc[k]; }
    const float inv = 1.f / sum;
#pragma unroll
    for (int k = 0; k < NB; ++k) s_attn[t * NB + k] = sc[k] * inv;
  }
  __syncthreads();

  for (int i = t; i < PM * KPAD; i += 256) {
    const int r2 = i / KPAD, c = i - r2 * KPAD;
    const int node = r2 >> 1;
    const float* aw = s_attn + r2 * NB;
    const unsigned short* col = s_kv + (node * NB) * KS + c;
    float s = 0.f;
#pragma unroll
    for (int k = 0; k < NB; ++k) s = fmaf(aw[k], bf2f(col[k * KS]), s);
    s_pool[r2 * KS + c] = f2bf(s);
  }
  __syncthreads();

  // ---- Phase D: out = pooled @ Wv^T (+bv), per-head row select ----
  for (int nt = wv; nt < 15; nt += 8) {
    v8f pacc[PMT];
#pragma unroll
    for (int pt = 0; pt < PMT; ++pt)
      pacc[pt] = (v8f){0.f, 0.f, 0.f, 0.f, 0.f, 0.f, 0.f, 0.f};
#pragma unroll
    for (int kt = 0; kt < KT; ++kt) {
      v16bf b = glb_b_frag(WvB, nt * 16, kt * 32, KPAD);
#pragma unroll
      for (int pt = 0; pt < PMT; ++pt) {
        v16bf a = lds_a_frag(s_pool, pt * 16, kt * 32, KS);
        pacc[pt] = wmma_bf16(a, b, pacc[pt]);
      }
    }
    const int n = nt * 16 + ln;
    const int head = (n < HDIM) ? 0 : 1;
    const float bvn = (n < QDIM) ? bv[n] : 0.f;
#pragma unroll
    for (int pt = 0; pt < PMT; ++pt) {
#pragma unroll
      for (int i = 0; i < 8; ++i) {
        const int r2 = pt * 16 + i + 8 * sel;     // pooled row = node*2 + h
        if ((r2 & 1) == head)
          s_out[(r2 >> 1) * 264 + n] = f2bf(pacc[pt][i] + bvn);
      }
    }
  }
  __syncthreads();

  // ---- Phase E: o2 = out @ Wo^T + bo ----
  for (int nt = wv; nt < 15; nt += 8) {
    v8f acc = {0.f, 0.f, 0.f, 0.f, 0.f, 0.f, 0.f, 0.f};
#pragma unroll
    for (int kt = 0; kt < 8; ++kt) {
      v16bf a = lds_a_frag(s_out, 0, kt * 32, 264);
      v16bf b = glb_b_frag(WoB, nt * 16, kt * 32, 256);
      acc = wmma_bf16(a, b, acc);
    }
    const int n = nt * 16 + ln;
    const float bon = (n < QDIM) ? bo[n] : 0.f;
#pragma unroll
    for (int i = 0; i < 8; ++i)
      s_o2[(i + 8 * sel) * 264 + n] = f2bf(acc[i] + bon);
  }
  __syncthreads();

  // ---- Phase F: h = relu(o2 @ Wm1^T + bm1) ----
  {
    const int nt = wv;                 // 8 tiles, one per wave
    v8f acc = {0.f, 0.f, 0.f, 0.f, 0.f, 0.f, 0.f, 0.f};
#pragma unroll
    for (int kt = 0; kt < 8; ++kt) {
      v16bf a = lds_a_frag(s_o2, 0, kt * 32, 264);
      v16bf b = glb_b_frag(Wm1B, nt * 16, kt * 32, 256);
      acc = wmma_bf16(a, b, acc);
    }
    const int n = nt * 16 + ln;
    const float bn = bm1[n];
#pragma unroll
    for (int i = 0; i < 8; ++i)
      s_h[(i + 8 * sel) * 136 + n] = f2bf(fmaxf(acc[i] + bn, 0.f));
  }
  __syncthreads();

  // ---- Phase G: res = h @ Wm2^T + bm2  -> global rows ----
  {
    const int nt = wv;
    v8f acc = {0.f, 0.f, 0.f, 0.f, 0.f, 0.f, 0.f, 0.f};
#pragma unroll
    for (int kt = 0; kt < 4; ++kt) {
      v16bf a = lds_a_frag(s_h, 0, kt * 32, 136);
      v16bf b = glb_b_frag(Wm2B, nt * 16, kt * 32, 128);
      acc = wmma_bf16(a, b, acc);
    }
    const int n = nt * 16 + ln;
    const float bn = bm2[n];
#pragma unroll
    for (int i = 0; i < 8; ++i) {
      const int nl = i + 8 * sel;
      if (nl < NODES) {
        const int gnode = node0 + nl;
        const int row = nids ? n2i[nids[gnode]] : gnode;
        dst[(size_t)row * 128 + n] = acc[i] + bn;
      }
    }
  }
}

// ---------------------------------------------------------------------------
// Link predictor: p = sigmoid(Wout . relu(Wsrc za + bsrc + Wdst zb + bdst) + bout)
// ---------------------------------------------------------------------------
__global__ __launch_bounds__(128) void predict_kernel(
    const float* __restrict__ z, const int* __restrict__ a_idx,
    const int* __restrict__ b_idx,
    const float* __restrict__ Wsrc, const float* __restrict__ bsrc,
    const float* __restrict__ Wdst, const float* __restrict__ bdst,
    const float* __restrict__ Wout, const float* __restrict__ bout,
    float* __restrict__ out)
{
  const int b = blockIdx.x, j = threadIdx.x;
  const float* za = z + (size_t)a_idx[b] * 128;
  const float* zb = z + (size_t)b_idx[b] * 128;
  float s = bsrc[j] + bdst[j];
  for (int i = 0; i < 128; ++i)
    s = fmaf(Wsrc[j * 128 + i], za[i], fmaf(Wdst[j * 128 + i], zb[i], s));
  __shared__ float red[128];
  red[j] = fmaxf(s, 0.f) * Wout[j];
  __syncthreads();
  for (int off = 64; off > 0; off >>= 1) {
    if (j < off) red[j] += red[j + off];
    __syncthreads();
  }
  if (j == 0) out[b] = 1.f / (1.f + __expf(-(red[0] + bout[0])));
}

// ---------------------------------------------------------------------------
extern "C" void kernel_launch(void* const* d_in, const int* in_sizes, int n_in,
                              void* d_out, int out_size, void* d_ws, size_t ws_size,
                              hipStream_t stream) {
  (void)n_in; (void)out_size; (void)ws_size;
  auto F = [&](int i) { return (const float*)d_in[i]; };
  auto I = [&](int i) { return (const int*)d_in[i]; };
  // d_in order = setup_inputs() insertion order (params dict recursed in order):
  // 0 time_w, 1 time_b,
  // 2..13  layer0: Wq,bq,Wk,bk,Wv,bv,Wo,bo,Wm1,bm1,Wm2,bm2
  // 14..25 layer1: same
  // 26 Wsrc 27 bsrc 28 Wdst 29 bdst 30 Wout 31 bout
  // 32 nids0 33 nids1 34 nid_to_idx 35 nbr_nids_idx0 36 times0 37 times1
  // 38 nbr_times0 39 nbr_times1 40 nbr_feats0 41 nbr_feats1 42 mask0 43 mask1
  // 44 src_idx 45 dst_idx 46 neg_idx
  const int N0 = in_sizes[32];
  const int N1 = in_sizes[33];
  const int U  = in_sizes[34];
  const int B  = in_sizes[44];

  char* ws = (char*)d_ws;
  size_t off = 0;
  auto alloc = [&](size_t bytes) -> char* {
    char* p = ws + off;
    off = (off + bytes + 255) & ~(size_t)255;
    return p;
  };
  float* z    = (float*)alloc((size_t)U * 128 * 4);
  float* out0 = (float*)alloc((size_t)N0 * 128 * 4);
  float* qbk  = (float*)alloc(4 * sizeof(float));
  unsigned short* wkB0  = (unsigned short*)alloc(16 * 320 * 2);
  unsigned short* wkB1  = (unsigned short*)alloc(16 * 192 * 2);
  unsigned short* WvB0  = (unsigned short*)alloc(240 * 320 * 2);
  unsigned short* WvB1  = (unsigned short*)alloc(240 * 192 * 2);
  unsigned short* WoB0  = (unsigned short*)alloc(240 * 256 * 2);
  unsigned short* WoB1  = (unsigned short*)alloc(240 * 256 * 2);
  unsigned short* Wm1B0 = (unsigned short*)alloc(128 * 256 * 2);
  unsigned short* Wm1B1 = (unsigned short*)alloc(128 * 256 * 2);
  unsigned short* Wm2B0 = (unsigned short*)alloc(128 * 128 * 2);
  unsigned short* Wm2B1 = (unsigned short*)alloc(128 * 128 * 2);

  prep1_kernel<<<1, 256, 0, stream>>>(F(0), F(1), F(2), F(3), F(4), F(5),
                                      F(14), F(15), F(16), F(17), qbk, wkB0, wkB1);

  auto pack = [&](const float* src, unsigned short* dst, int n_src, int ks,
                  int koff, int kw, int Nd, int Kd) {
    const int total = Nd * Kd;
    pack_w_kernel<<<(total + 255) / 256, 256, 0, stream>>>(src, dst, n_src, ks,
                                                           koff, kw, Nd, Kd);
  };
  pack(F(6),  WvB0, 228, 292, 0,   292, 240, 320);   // layer0 Wv: identity cols
  pack(F(18), WvB1, 228, 292, 128, 164, 240, 192);   // layer1 Wv: skip zero nbr cols
  pack(F(8),  WoB0, 228, 228, 0, 228, 240, 256);
  pack(F(20), WoB1, 228, 228, 0, 228, 240, 256);
  pack(F(10), Wm1B0, 128, 356, 0, 228, 128, 256);    // node_feat part of Wm1 is x0
  pack(F(22), Wm1B1, 128, 356, 0, 228, 128, 256);
  pack(F(12), Wm2B0, 128, 128, 0, 128, 128, 128);
  pack(F(24), Wm2B1, 128, 128, 0, 128, 128, 128);

  const int n4 = U * 128 / 4;
  zero_f4_kernel<<<(n4 + 255) / 256, 256, 0, stream>>>((float4*)z, n4);

  // Layer 1: 49152 nodes, kv = [edge(64), time(100)] -> Kpad 192. Writes z rows.
  attn_fused_kernel<16, 192, 164, false><<<N1 / 16, 256, 0, stream>>>(
      F(41), F(37), F(39), nullptr, nullptr, I(33), I(34), F(0), F(1),
      WvB1, wkB1, WoB1, Wm1B1, Wm2B1, qbk + 2, F(19), F(21), F(23), F(25), z);

  // Layer 0: 3072 nodes, kv = [z gather(128), edge(64), time(100)] -> Kpad 320.
  // Writes to out0 (z must stay readable until done), then scatter.
  attn_fused_kernel<8, 320, 292, true><<<N0 / 8, 256, 0, stream>>>(
      F(40), F(36), F(38), I(35), z, nullptr, nullptr, F(0), F(1),
      WvB0, wkB0, WoB0, Wm1B0, Wm2B0, qbk + 0, F(7), F(9), F(11), F(13), out0);
  scatter_rows_kernel<<<(N0 * 128 + 255) / 256, 256, 0, stream>>>(out0, z, I(32),
                                                                  I(34), N0);

  float* out = (float*)d_out;
  predict_kernel<<<B, 128, 0, stream>>>(z, I(44), I(45), F(26), F(27), F(28),
                                        F(29), F(30), F(31), out);
  predict_kernel<<<B, 128, 0, stream>>>(z, I(44), I(46), F(26), F(27), F(28),
                                        F(29), F(30), F(31), out + B);
}
